// LoRAMultiheadAttention_35588099015583
// MI455X (gfx1250) — matrix operand order repaired and checked by reading, not compiled, not run on hardware
//
#include <hip/hip_runtime.h>
#include <hip/hip_bf16.h>
#include <float.h>

// ---------------- problem constants ----------------
#define Lseq 2048
#define Nb   2
#define Edim 1024
#define Hh   16
#define HD   64
#define Rr   8
#define Mtok (Lseq * Nb)       // 4096
#define SCALING 4.0f           // ALPHA / R

typedef __attribute__((ext_vector_type(16))) _Float16 v16h;
typedef __attribute__((ext_vector_type(8)))  float    v8f;

// ---------------- WMMA helpers ----------------
__device__ __forceinline__ v8f wmma_f16(v16h a, v16h b, v8f c) {
  // D = A(16x32 f16) * B(32x16 f16) + C(16x16 f32)
  return __builtin_amdgcn_wmma_f32_16x16x32_f16(
      /*neg_a=*/false, a, /*neg_b=*/false, b,
      /*c_mod=*/(short)0, c, /*reuse_a=*/false, /*reuse_b=*/false);
}

// A-matrix 16x32 f16 fragment.  Source: row-major tile, leading dim ld.
// lane: m = lane&15, half = lane>>4
//   elems[0..7]  = src[m][half*8 + 0..7]
//   elems[8..15] = src[m][16 + half*8 + 0..7]
__device__ __forceinline__ v16h load_a_frag(const _Float16* __restrict__ tile, int ld) {
  const int lane = threadIdx.x & 31;
  const int m = lane & 15, hf = lane >> 4;
  const _Float16* p = tile + (size_t)m * ld;
  v16h r;
#pragma unroll
  for (int e = 0; e < 8; ++e) r[e] = p[hf * 8 + e];
#pragma unroll
  for (int e = 0; e < 8; ++e) r[8 + e] = p[16 + hf * 8 + e];
  return r;
}

// B-matrix 32x16 f16 fragment, B[k][n].  Source laid out "column-major":
// src row index = n (the B column), k contiguous within the row, leading dim ld.
// lane: n = lane&15, half = lane>>4 ; elems[0..7] = src[n][half*8 + 0..7], elems[8..15] = src[n][16 + half*8 + 0..7]
__device__ __forceinline__ v16h load_b_frag(const _Float16* __restrict__ tile, int ld) {
  const int lane = threadIdx.x & 31;
  const int nn = lane & 15, hf = lane >> 4;
  const _Float16* p = tile + (size_t)nn * ld;
  v16h r;
#pragma unroll
  for (int e = 0; e < 8; ++e) r[e] = p[hf * 8 + e];
#pragma unroll
  for (int e = 0; e < 8; ++e) r[8 + e] = p[16 + hf * 8 + e];
  return r;
}

// ---------------- elementwise f32 -> f16 ----------------
__global__ void cvt_f32_f16_kernel(const float* __restrict__ src,
                                   _Float16* __restrict__ dst, size_t n) {
  size_t i = (size_t)blockIdx.x * blockDim.x + threadIdx.x;
  size_t stride = (size_t)gridDim.x * blockDim.x;
  for (; i < n; i += stride) dst[i] = (_Float16)src[i];
}

// ---------------- LoRA rank projection: T[m][r] = sum_i X[m][i] * A[r][i] ----------------
__global__ __launch_bounds__(32) void lora_t_kernel(const _Float16* __restrict__ X,
                                                    const float* __restrict__ A,
                                                    float* __restrict__ T, int K) {
  const int m = blockIdx.x;
  const int r = blockIdx.y;
  const int lane = threadIdx.x;
  float s = 0.f;
  const _Float16* xr = X + (size_t)m * K;
  const float* ar = A + (size_t)r * K;
  for (int i = lane; i < K; i += 32) s += (float)xr[i] * ar[i];
#pragma unroll
  for (int off = 16; off > 0; off >>= 1) s += __shfl_down(s, off, 32);
  if (lane == 0) T[(size_t)m * Rr + r] = s;
}

// ---------------- fused GEMM + bias + LoRA epilogue ----------------
// out[m][o] = sum_i Xh[m][i]*Wh[o][i] + bias[o] + SCALING * sum_r T[m][r]*Bp[o][r]
// mode 0: f16 store into (n,h,L,HD) head layout, value *= scale
// mode 1: f16 store into (n,h,HD,L) transposed head layout (for V)
// mode 2: fp32 store into row-major [m][o] (== (L,N,E) flat)
__global__ __launch_bounds__(32) void gemm_lora_kernel(
    const _Float16* __restrict__ Xh,   // M x K
    const _Float16* __restrict__ Wh,   // Eo x K (out-major)
    const float* __restrict__ bias,    // Eo
    const float* __restrict__ T,       // M x R
    const float* __restrict__ Bp,      // Eo x R
    _Float16* __restrict__ out16, float* __restrict__ out32,
    int K, int mode, float scale) {
  const int lane = threadIdx.x & 31;
  const int nt = lane & 15, hf = lane >> 4;
  const int row0 = blockIdx.y * 16;
  const int col0 = blockIdx.x * 64;

  v8f acc[4] = {};
  for (int k0 = 0; k0 < K; k0 += 32) {
    v16h a = load_a_frag(Xh + (size_t)row0 * K + k0, K);
#pragma unroll
    for (int dt = 0; dt < 4; ++dt) {
      v16h b = load_b_frag(Wh + (size_t)(col0 + dt * 16) * K + k0, K);
      acc[dt] = wmma_f16(a, b, acc[dt]);
    }
  }

  // per-lane LoRA t rows (8 rows owned by this lane)
  float tl[8][Rr];
#pragma unroll
  for (int j = 0; j < 8; ++j) {
    int m = row0 + j + 8 * hf;
#pragma unroll
    for (int r = 0; r < Rr; ++r) tl[j][r] = T[(size_t)m * Rr + r];
  }

#pragma unroll
  for (int dt = 0; dt < 4; ++dt) {
    const int o = col0 + dt * 16 + nt;
    const float bo = bias[o];
    float br[Rr];
#pragma unroll
    for (int r = 0; r < Rr; ++r) br[r] = Bp[(size_t)o * Rr + r];
#pragma unroll
    for (int j = 0; j < 8; ++j) {
      const int m = row0 + j + 8 * hf;
      float lora = 0.f;
#pragma unroll
      for (int r = 0; r < Rr; ++r) lora += tl[j][r] * br[r];
      float val = (acc[dt][j] + bo + SCALING * lora) * scale;
      if (mode == 2) {
        out32[(size_t)m * Edim + o] = val;
      } else {
        const int ls = m / Nb, nbatch = m % Nb;
        const int h = o >> 6, d = o & 63;
        if (mode == 0)
          out16[(((size_t)(nbatch * Hh + h)) * Lseq + ls) * HD + d] = (_Float16)val;
        else  // transposed V: (n,h,HD,L)
          out16[(((size_t)(nbatch * Hh + h)) * HD + d) * Lseq + ls] = (_Float16)val;
      }
    }
  }
}

// ---------------- attention: one wave per (q-tile16, h, n) ----------------
__global__ __launch_bounds__(32) void attn_kernel(
    const _Float16* __restrict__ qh,   // (n,h,L,HD), q pre-scaled
    const _Float16* __restrict__ kh,   // (n,h,L,HD)
    const _Float16* __restrict__ vt,   // (n,h,HD,L)
    const int* __restrict__ mask,      // (n,L) nonzero = masked
    float* __restrict__ wout,          // (L,N,L) fp32 (pre-zeroed; atomic mean accum)
    _Float16* __restrict__ aout) {     // (M,E) f16 merged heads
  __shared__ _Float16 wlds[16 * 32];
  const int lane = threadIdx.x & 31;
  const int nt = lane & 15, hf = lane >> 4;
  const int q0 = blockIdx.x * 16;
  const int h = blockIdx.y;
  const int n = blockIdx.z;

  const _Float16* qp = qh + ((size_t)(n * Hh + h) * Lseq) * HD;
  const _Float16* kp = kh + ((size_t)(n * Hh + h) * Lseq) * HD;
  const _Float16* vp = vt + ((size_t)(n * Hh + h) * HD) * Lseq;
  const int* mp = mask + (size_t)n * Lseq;

  const v16h aq0 = load_a_frag(qp + (size_t)q0 * HD, HD);
  const v16h aq1 = load_a_frag(qp + (size_t)q0 * HD + 32, HD);

  float rmax[8], rsum[8];
#pragma unroll
  for (int j = 0; j < 8; ++j) { rmax[j] = -3.0e38f; rsum[j] = 0.f; }

  // ---- pass 1: online row max / sum ----
  for (int kt = 0; kt < Lseq / 16; ++kt) {
    v8f c = {};
    v16h b0 = load_b_frag(kp + (size_t)kt * 16 * HD, HD);
    v16h b1 = load_b_frag(kp + (size_t)kt * 16 * HD + 32, HD);
    c = wmma_f16(aq0, b0, c);
    c = wmma_f16(aq1, b1, c);
    const bool km = mp[kt * 16 + nt] != 0;
#pragma unroll
    for (int j = 0; j < 8; ++j) {
      float tm = km ? -3.0e38f : c[j];
#pragma unroll
      for (int off = 1; off < 16; off <<= 1)
        tm = fmaxf(tm, __shfl_xor(tm, off, 32));
      const float mn = fmaxf(rmax[j], tm);
      float ex = km ? 0.f : __expf(c[j] - mn);
#pragma unroll
      for (int off = 1; off < 16; off <<= 1) ex += __shfl_xor(ex, off, 32);
      rsum[j] = rsum[j] * __expf(rmax[j] - mn) + ex;
      rmax[j] = mn;
    }
  }

  float rscale[8];
#pragma unroll
  for (int j = 0; j < 8; ++j) rscale[j] = 1.0f / fmaxf(rsum[j], FLT_MIN);

  // ---- pass 2: recompute, emit weights, accumulate P @ V ----
  v8f oacc[4] = {};
  const float wf = 1.0f / (float)Hh;
  for (int kt2 = 0; kt2 < Lseq / 32; ++kt2) {
#pragma unroll
    for (int sub = 0; sub < 2; ++sub) {
      const int kt = kt2 * 2 + sub;
      v8f c = {};
      v16h b0 = load_b_frag(kp + (size_t)kt * 16 * HD, HD);
      v16h b1 = load_b_frag(kp + (size_t)kt * 16 * HD + 32, HD);
      c = wmma_f16(aq0, b0, c);
      c = wmma_f16(aq1, b1, c);
      const bool km = mp[kt * 16 + nt] != 0;
#pragma unroll
      for (int j = 0; j < 8; ++j) {
        const float w = km ? 0.f : __expf(c[j] - rmax[j]) * rscale[j];
        const int qg = q0 + j + 8 * hf;
        atomicAdd(&wout[((size_t)qg * Nb + n) * Lseq + kt * 16 + nt], w * wf);
        wlds[(j + 8 * hf) * 32 + sub * 16 + nt] = (_Float16)w;
      }
    }
    __syncthreads();
    const v16h aw = load_a_frag(wlds, 32);
#pragma unroll
    for (int dt = 0; dt < 4; ++dt) {
      v16h bv = load_b_frag(vp + (size_t)(dt * 16) * Lseq + kt2 * 32, Lseq);
      oacc[dt] = wmma_f16(aw, bv, oacc[dt]);
    }
    __syncthreads();
  }

  // store merged-head tile: aout[(qg*Nb+n)][h*64+d]
#pragma unroll
  for (int dt = 0; dt < 4; ++dt) {
    const int d = dt * 16 + nt;
#pragma unroll
    for (int j = 0; j < 8; ++j) {
      const int qg = q0 + j + 8 * hf;
      aout[((size_t)qg * Nb + n) * Edim + h * HD + d] = (_Float16)oacc[dt][j];
    }
  }
}

// ---------------- host launcher ----------------
extern "C" void kernel_launch(void* const* d_in, const int* in_sizes, int n_in,
                              void* d_out, int out_size, void* d_ws, size_t ws_size,
                              hipStream_t stream) {
  (void)in_sizes; (void)n_in; (void)out_size; (void)ws_size;
  const float* x    = (const float*)d_in[0];   // (L,N,E)
  const int*   mask = (const int*)d_in[1];     // (N,L)
  const float* Wp[4] = {(const float*)d_in[2],  (const float*)d_in[6],
                        (const float*)d_in[10], (const float*)d_in[14]};
  const float* bp[4] = {(const float*)d_in[3],  (const float*)d_in[7],
                        (const float*)d_in[11], (const float*)d_in[15]};
  const float* Ap[4] = {(const float*)d_in[4],  (const float*)d_in[8],
                        (const float*)d_in[12], (const float*)d_in[16]};
  const float* Bp[4] = {(const float*)d_in[5],  (const float*)d_in[9],
                        (const float*)d_in[13], (const float*)d_in[17]};

  float* attn_output = (float*)d_out;                              // L*N*E
  float* wmean = attn_output + (size_t)Lseq * Nb * Edim;           // L*N*L

  // workspace layout
  char* ws = (char*)d_ws;
  size_t off = 0;
  auto walloc = [&](size_t bytes) -> void* {
    void* p = ws + off; off += (bytes + 255) & ~(size_t)255; return p;
  };
  _Float16* Xh    = (_Float16*)walloc((size_t)Mtok * Edim * 2);
  _Float16* Wh[4];
  for (int p = 0; p < 4; ++p) Wh[p] = (_Float16*)walloc((size_t)Edim * Edim * 2);
  _Float16* qhb   = (_Float16*)walloc((size_t)Nb * Hh * Lseq * HD * 2);
  _Float16* khb   = (_Float16*)walloc((size_t)Nb * Hh * Lseq * HD * 2);
  _Float16* vtb   = (_Float16*)walloc((size_t)Nb * Hh * HD * Lseq * 2);
  _Float16* aouth = (_Float16*)walloc((size_t)Mtok * Edim * 2);
  float*    Tbuf[4];
  for (int p = 0; p < 4; ++p) Tbuf[p] = (float*)walloc((size_t)Mtok * Rr * 4);

  // 1) f32 -> f16 conversions
  cvt_f32_f16_kernel<<<1024, 256, 0, stream>>>(x, Xh, (size_t)Mtok * Edim);
  for (int p = 0; p < 4; ++p)
    cvt_f32_f16_kernel<<<1024, 256, 0, stream>>>(Wp[p], Wh[p], (size_t)Edim * Edim);

  // 2) LoRA rank projections for q,k,v
  for (int p = 0; p < 3; ++p)
    lora_t_kernel<<<dim3(Mtok, Rr), 32, 0, stream>>>(Xh, Ap[p], Tbuf[p], Edim);

  // 3) q,k,v projections (WMMA) with fused bias+LoRA epilogue
  const dim3 ggrid(Edim / 64, Mtok / 16);
  const float qscale = 0.125f;  // HD^-0.5
  gemm_lora_kernel<<<ggrid, 32, 0, stream>>>(Xh, Wh[0], bp[0], Tbuf[0], Bp[0],
                                             qhb, nullptr, Edim, 0, qscale);
  gemm_lora_kernel<<<ggrid, 32, 0, stream>>>(Xh, Wh[1], bp[1], Tbuf[1], Bp[1],
                                             khb, nullptr, Edim, 0, 1.0f);
  gemm_lora_kernel<<<ggrid, 32, 0, stream>>>(Xh, Wh[2], bp[2], Tbuf[2], Bp[2],
                                             vtb, nullptr, Edim, 1, 1.0f);

  // 4) zero the attention-weights output region (atomics accumulate the head mean)
  hipMemsetAsync(wmean, 0, (size_t)Lseq * Nb * Lseq * sizeof(float), stream);

  // 5) flash-style attention (WMMA QK^T and PV, f32 softmax, atomic weight mean)
  attn_kernel<<<dim3(Lseq / 16, Hh, Nb), 32, 0, stream>>>(qhb, khb, vtb, mask,
                                                          wmean, aouth);

  // 6) output projection with LoRA
  lora_t_kernel<<<dim3(Mtok, Rr), 32, 0, stream>>>(aouth, Ap[3], Tbuf[3], Edim);
  gemm_lora_kernel<<<ggrid, 32, 0, stream>>>(aouth, Wh[3], bp[3], Tbuf[3], Bp[3],
                                             nullptr, attn_output, Edim, 2, 1.0f);
}
